// VqrgbNet_23845658427573
// MI455X (gfx1250) — compile-verified, matrix operand order checked by reading, not run on hardware
//
#include <hip/hip_runtime.h>

typedef __attribute__((ext_vector_type(16))) __bf16 v16bf;
typedef __attribute__((ext_vector_type(8)))  __bf16 v8bf;
typedef __attribute__((ext_vector_type(8)))  float  v8f;
typedef __attribute__((ext_vector_type(4)))  unsigned v4u;

__device__ __forceinline__ __bf16 f2bf(float f) {
  unsigned u = __builtin_bit_cast(unsigned, f);
  unsigned r = (u + 0x7FFFu + ((u >> 16) & 1u)) >> 16;   // RNE
  unsigned short h = (unsigned short)r;
  return __builtin_bit_cast(__bf16, h);
}
__device__ __forceinline__ float bf2f(__bf16 b) {
  unsigned short h = __builtin_bit_cast(unsigned short, b);
  return __builtin_bit_cast(float, ((unsigned)h) << 16);
}
__device__ __forceinline__ v4u zero4() { v4u z; z[0]=0u; z[1]=0u; z[2]=0u; z[3]=0u; return z; }
// relu on 8 packed bf16 (sign bit -> zero that half)
__device__ __forceinline__ v4u relu_pk(v4u u4) {
#pragma unroll
  for (int x = 0; x < 4; ++x) {
    unsigned u  = u4[x];
    unsigned lo = (u & 0x00008000u) ? 0u : (u & 0x0000FFFFu);
    unsigned hi = (u & 0x80000000u) ? 0u : (u & 0xFFFF0000u);
    u4[x] = lo | hi;
  }
  return u4;
}
// (kh,kw) from slice index s (s < 16), KW in {1,3,4}; uniform branch
__device__ __forceinline__ void kdec(int s, int KW, int& kh, int& kw) {
  if (KW == 4)      { kh = s >> 2;        kw = s & 3; }
  else if (KW == 3) { kh = (s * 11) >> 5; kw = s - kh * 3; }   // exact, s<16
  else              { kh = s;             kw = 0; }
}

// ---------------------------------------------------------------------------
// Implicit-GEMM conv / convT on v_wmma_f32_16x16x32_bf16.
// Activations: bf16 NHWC, channels padded to Cin_p = 1<<cinp_sh (multiple of 8).
// Weights: pre-packed bf16 [Cout_p][Kpad], K ordered (kh,kw,cin-padded).
// Hout*Wout = 1<<hwo_sh, Wout = 1<<wout_sh (all spatial sizes are pow2).
// out_mode: 0 = bf16 NHWC, 1 = fp32 NHWC (VQ scores), 2 = fp32 NCHW (+MSE loss)
// ---------------------------------------------------------------------------
template <int MT>
__global__ __launch_bounds__(32) void conv_wmma_kernel(
    void* __restrict__ out, const __bf16* __restrict__ in,
    const __bf16* __restrict__ Wp, const float* __restrict__ bias,
    const __bf16* __restrict__ resid, const float* __restrict__ mnorm,
    const float* __restrict__ lossref, float* __restrict__ lossacc,
    int N, int Hin, int Win, int hwo_sh, int wout_sh, int Cout, int Cout_p,
    int cinp_sh, int KH, int KW, int Kpad, int stride, int pad,
    int transposed, int in_relu, int out_relu, int out_mode)
{
  const int lane = threadIdx.x;
  const int half = lane >> 4;
  const int ln   = lane & 15;
  const int HWo  = 1 << hwo_sh;
  const int Wout = 1 << wout_sh;
  const int Ntot = N << hwo_sh;
  const int khkw = KH * KW;

  const int ntile0 = blockIdx.x * 16;
  const int m0     = blockIdx.y * (16 * MT);

  const int n = ntile0 + ln;
  const bool nvalid = n < Ntot;
  const int img = n >> hwo_sh;
  const int hw  = n & (HWo - 1);
  const int oh  = hw >> wout_sh;
  const int ow  = hw & (Wout - 1);
  // hoisted bases: normal conv uses oh*stride-pad; transposed uses oh-pad
  const int ihb = transposed ? (oh - pad) : (oh * stride - pad);
  const int iwb = transposed ? (ow - pad) : (ow * stride - pad);

  v8f acc[MT];
#pragma unroll
  for (int t = 0; t < MT; ++t)
#pragma unroll
    for (int r = 0; r < 8; ++r) acc[t][r] = 0.f;

  for (int kb = 0; kb < Kpad; kb += 32) {
    // ---- B fragment: two b128 runs of 8 consecutive K ----
    v4u raw0 = zero4(), raw1 = zero4();
    if (cinp_sh >= 5) {
      // both runs live in the same (kh,kw) slice: one decode, two loads
      int k8 = kb + half * 8;
      int s  = k8 >> cinp_sh;
      int c0 = k8 & ((1 << cinp_sh) - 1);
      if (nvalid && s < khkw) {
        int kh, kw; kdec(s, KW, kh, kw);
        int ih, iw; bool ok;
        if (transposed) {                    // lhs_dilation = 2
          int ihd = ihb + kh, iwd = iwb + kw;
          ok = (ihd >= 0) && (iwd >= 0) && !(ihd & 1) && !(iwd & 1);
          ih = ihd >> 1; iw = iwd >> 1;
          ok = ok && (ih < Hin) && (iw < Win);
        } else {
          ih = ihb + kh; iw = iwb + kw;
          ok = (ih >= 0) && (ih < Hin) && (iw >= 0) && (iw < Win);
        }
        if (ok) {
          const __bf16* p = in + ((((img * Hin + ih) * Win + iw) << cinp_sh) + c0);
          raw0 = *(const v4u*)(p);
          raw1 = *(const v4u*)(p + 16);
          if (in_relu) { raw0 = relu_pk(raw0); raw1 = relu_pk(raw1); }
        }
      }
    } else {                                 // enc_c1: Cin_p = 8, per-run decode
#pragma unroll
      for (int j = 0; j < 2; ++j) {
        int k8 = kb + half * 8 + j * 16;
        int s  = k8 >> cinp_sh;
        int c0 = k8 & ((1 << cinp_sh) - 1);
        if (nvalid && s < khkw) {
          int kh, kw; kdec(s, KW, kh, kw);
          int ih = ihb + kh, iw = iwb + kw;
          if (ih >= 0 && ih < Hin && iw >= 0 && iw < Win) {
            v4u r4 = *(const v4u*)(in + ((((img * Hin + ih) * Win + iw) << cinp_sh) + c0));
            if (in_relu) r4 = relu_pk(r4);
            if (j == 0) raw0 = r4; else raw1 = r4;
          }
        }
      }
    }
    v8bf b0 = __builtin_bit_cast(v8bf, raw0);
    v8bf b1 = __builtin_bit_cast(v8bf, raw1);
    v16bf bfrag = __builtin_shufflevector(b0, b1,
        0,1,2,3,4,5,6,7,8,9,10,11,12,13,14,15);

    // ---- A fragments: packed weight rows, 2 b128 loads per tile ----
    const int u0 = (kb >> 3) + half;         // v8bf units
#pragma unroll
    for (int t = 0; t < MT; ++t) {
      const v8bf* wr = (const v8bf*)(Wp + (size_t)(m0 + t * 16 + ln) * Kpad);
      v8bf a0 = wr[u0];
      v8bf a1 = wr[u0 + 2];
      v16bf afrag = __builtin_shufflevector(a0, a1,
          0,1,2,3,4,5,6,7,8,9,10,11,12,13,14,15);
      acc[t] = __builtin_amdgcn_wmma_f32_16x16x32_bf16(
          false, afrag, false, bfrag, (short)0, acc[t], false, false);
    }
  }

  // ---- epilogue ----
  float lsum = 0.f;
  if (nvalid) {
    const int nhwc = ((img << hwo_sh) + hw) * Cout_p;
#pragma unroll
    for (int t = 0; t < MT; ++t) {
      const int mb = m0 + t * 16 + half * 8;
      if (out_mode == 2) {                   // fp32 NCHW (+ loss)
        float* fo = (float*)out;
#pragma unroll
        for (int r = 0; r < 8; ++r) {
          int m = mb + r;
          if (m < Cout) {
            float val = acc[t][r];
            if (bias) val += bias[m];
            if (out_relu) val = fmaxf(val, 0.f);
            int idx = ((img * Cout + m) << hwo_sh) + hw;
            fo[idx] = val;
            if (lossref) { float d = lossref[idx] - val; lsum += d * d; }
          }
        }
      } else {
        v8bf rsd;
        if (resid) rsd = *(const v8bf*)(resid + nhwc + mb);
        float vals[8];
#pragma unroll
        for (int r = 0; r < 8; ++r) {
          int m = mb + r;
          float val = acc[t][r];
          if (mnorm) val = mnorm[m] - 2.0f * val;     // ||e||^2 - 2 z.e
          if (bias)  val += bias[m];
          if (resid) val += bf2f(rsd[r]);
          if (out_relu) val = fmaxf(val, 0.f);
          vals[r] = val;
        }
        if (out_mode == 1) {                 // fp32 NHWC (scores)
          float* fo = (float*)out + nhwc + mb;
          float4 f0 = make_float4(vals[0], vals[1], vals[2], vals[3]);
          float4 f1 = make_float4(vals[4], vals[5], vals[6], vals[7]);
          *(float4*)(fo)     = f0;
          *(float4*)(fo + 4) = f1;
        } else {                             // bf16 NHWC
          v8bf ov;
#pragma unroll
          for (int r = 0; r < 8; ++r) ov[r] = f2bf(vals[r]);
          *(v8bf*)((__bf16*)out + nhwc + mb) = ov;
        }
      }
    }
  }
  if (lossacc) {
#pragma unroll
    for (int off = 16; off > 0; off >>= 1) lsum += __shfl_down(lsum, off, 32);
    if (lane == 0) atomicAdd(lossacc, lsum);
  }
}

// ---- pack rgb fp32 NCHW [16,3,256,256] -> bf16 NHWC [16,256,256,8] ----
__global__ void pack_input_kernel(const float* __restrict__ src,
                                  __bf16* __restrict__ dst) {
  int v = blockIdx.x * blockDim.x + threadIdx.x;   // 16*65536
  if (v >= 16 * 65536) return;
  int img = v >> 16, hw = v & 65535;
  v8bf o;
#pragma unroll
  for (int c = 0; c < 8; ++c)
    o[c] = (c < 3) ? f2bf(src[(img * 3 + c) * 65536 + hw]) : f2bf(0.f);
  *(v8bf*)(dst + (size_t)v * 8) = o;
}

// ---- pack conv weights -> bf16 [Cout_p][Kpad], K=(kh,kw,cin-padded) ----
__global__ void pack_weight_kernel(const float* __restrict__ src,
    __bf16* __restrict__ dst, int Cout, int Cin, int cinp_sh,
    int KH, int KW, int Kpad, int transposed, int total) {
  int i = blockIdx.x * 256 + threadIdx.x;
  if (i >= total) return;
  int m = i / Kpad, k = i - m * Kpad;
  int s = k >> cinp_sh, cin = k & ((1 << cinp_sh) - 1);
  int khkw = KH * KW;
  float v = 0.f;
  if (m < Cout && s < khkw && cin < Cin) {
    int kh = s / KW, kw = s - kh * KW;
    if (transposed)   // wk[m][cin][kh][kw] = w[cin][m][KH-1-kh][KW-1-kw]
      v = src[((cin * Cout + m) * KH + (KH - 1 - kh)) * KW + (KW - 1 - kw)];
    else
      v = src[(m * Cin + cin) * khkw + kh * KW + kw];
  }
  dst[i] = f2bf(v);
}

__global__ void code_norms_kernel(const float* __restrict__ E,
                                  float* __restrict__ norms) {
  int c = blockIdx.x * blockDim.x + threadIdx.x;
  if (c < 512) {
    float s = 0.f;
    for (int d = 0; d < 64; ++d) { float v = E[c * 64 + d]; s += v * v; }
    norms[c] = s;
  }
}

// scores fp32 NHWC [16,32,32,512]: row of 512 contiguous per vector
__global__ __launch_bounds__(32) void argmin_kernel(
    const float* __restrict__ S, int* __restrict__ inds,
    float* __restrict__ ind_out) {
  int v = blockIdx.x;
  int lane = threadIdx.x;
  float best = 3.4e38f; int bi = 0x7fffffff;
  for (int c = lane; c < 512; c += 32) {
    float s = S[v * 512 + c];
    if (s < best || (s == best && c < bi)) { best = s; bi = c; }
  }
#pragma unroll
  for (int off = 16; off > 0; off >>= 1) {
    float ob = __shfl_down(best, off, 32);
    int   oi = __shfl_down(bi, off, 32);
    if (ob < best || (ob == best && oi < bi)) { best = ob; bi = oi; }
  }
  if (lane == 0) { inds[v] = bi; ind_out[v] = (float)bi; }
}

// z,q bf16 NHWC (64 contiguous per vector); latent MSE accumulation
__global__ __launch_bounds__(64) void gather_latent_kernel(
    const __bf16* __restrict__ z, const float* __restrict__ E,
    const int* __restrict__ inds, __bf16* __restrict__ q,
    float* __restrict__ latacc) {
  int v = blockIdx.x, d = threadIdx.x;
  int code = inds[v];
  float qf = E[code * 64 + d];
  float diff = qf - bf2f(z[v * 64 + d]);
  q[v * 64 + d] = f2bf(qf);
  float s = diff * diff;
#pragma unroll
  for (int off = 16; off > 0; off >>= 1) s += __shfl_down(s, off, 32);
  if ((d & 31) == 0) atomicAdd(latacc, s);
}

__global__ void finalize_kernel(const float* __restrict__ acc,
                                float* __restrict__ out0) {
  float recon  = acc[0] * (1.0f / (16.f * 3.f * 256.f * 256.f));
  float latent = 1.25f * acc[1] * (1.0f / (16.f * 1024.f * 64.f));
  out0[0] = recon + latent;
}

// ---------------------------------------------------------------------------
static inline void conv_launch(hipStream_t s, void* out, const __bf16* in,
    const __bf16* Wp, const float* bias, const __bf16* resid,
    const float* mnorm, const float* lossref, float* lossacc,
    int N, int Hin, int Win, int hwo_sh, int wout_sh, int Cout, int Cout_p,
    int cinp_sh, int KH, int KW, int Kpad, int stride, int pad,
    int transposed, int in_relu, int out_relu, int out_mode) {
  int Ntiles = ((N << hwo_sh) + 15) / 16;
  if (Cout_p >= 64) {
    conv_wmma_kernel<4><<<dim3(Ntiles, Cout_p / 64), 32, 0, s>>>(
        out, in, Wp, bias, resid, mnorm, lossref, lossacc,
        N, Hin, Win, hwo_sh, wout_sh, Cout, Cout_p, cinp_sh, KH, KW, Kpad,
        stride, pad, transposed, in_relu, out_relu, out_mode);
  } else {
    conv_wmma_kernel<1><<<dim3(Ntiles, Cout_p / 16), 32, 0, s>>>(
        out, in, Wp, bias, resid, mnorm, lossref, lossacc,
        N, Hin, Win, hwo_sh, wout_sh, Cout, Cout_p, cinp_sh, KH, KW, Kpad,
        stride, pad, transposed, in_relu, out_relu, out_mode);
  }
}

extern "C" void kernel_launch(void* const* d_in, const int* in_sizes, int n_in,
                              void* d_out, int out_size, void* d_ws, size_t ws_size,
                              hipStream_t stream) {
  (void)in_sizes; (void)n_in; (void)out_size; (void)ws_size;
  const float* rgb = (const float*)d_in[0];
  const float* cbk = (const float*)d_in[17];

  char* ws = (char*)d_ws;
  const size_t MB = 1ull << 20;
  __bf16* P0  = (__bf16*)(ws);             // 32 MB  x1 / y2
  float*  SCR = (float*) (ws + 32 * MB);   // 32 MB  VQ scores (fp32 NHWC)
  __bf16* INP = (__bf16*)(ws + 32 * MB);   // 16 MB  packed rgb (dead before SCR)
  __bf16* P2  = (__bf16*)(ws + 64 * MB);   // 16 MB  x2 / y1
  __bf16* P3  = (__bf16*)(ws + 80 * MB);   // 4 MB   x3 / dec stack
  __bf16* P4  = (__bf16*)(ws + 84 * MB);   // 4 MB   enc stack
  __bf16* P5  = (__bf16*)(ws + 88 * MB);   // 2 MB   res hidden (64ch)
  __bf16* P6  = (__bf16*)(ws + 90 * MB);   // 2 MB   z
  __bf16* P7  = (__bf16*)(ws + 92 * MB);   // 2 MB   quantized
  int*    IND = (int*)   (ws + 94 * MB);
  float*  NRM = (float*) (ws + 94 * MB + 0x10000);
  float*  ACC = (float*) (ws + 94 * MB + 0x11000);   // [recon, latent]
  __bf16* WPK = (__bf16*)(ws + 95 * MB);   // packed weights (~3 MB)

  // ---- weight packing ----
  size_t wcur = 0;
  auto pack = [&](int pi, int Cout, int Cout_p, int Cin, int cinp_sh,
                  int KH, int KW, int transposed) -> __bf16* {
    int Kpad = ((KH * KW << cinp_sh) + 31) & ~31;
    __bf16* dst = WPK + wcur;
    int total = Cout_p * Kpad;
    pack_weight_kernel<<<(total + 255) / 256, 256, 0, stream>>>(
        (const float*)d_in[pi], dst, Cout, Cin, cinp_sh, KH, KW, Kpad,
        transposed, total);
    wcur += (size_t)total;
    return dst;
  };
  __bf16* Wec1 = pack(1, 64, 64, 3, 3, 4, 4, 0);
  __bf16* Wec2 = pack(3, 128, 128, 64, 6, 4, 4, 0);
  __bf16* Wec3 = pack(5, 128, 128, 128, 7, 4, 4, 0);
  __bf16* Wec4 = pack(7, 128, 128, 128, 7, 3, 3, 0);
  __bf16* Wer1[3], *Wer2[3], *Wdr1[3], *Wdr2[3];
  for (int i = 0; i < 3; ++i) {
    Wer1[i] = pack(9 + 2 * i, 64, 64, 128, 7, 3, 3, 0);
    Wer2[i] = pack(10 + 2 * i, 128, 128, 64, 6, 1, 1, 0);
  }
  __bf16* Wpvq = pack(15, 64, 64, 128, 7, 1, 1, 0);
  __bf16* Wcbk = pack(17, 512, 512, 64, 6, 1, 1, 0);
  __bf16* Wdc1 = pack(18, 128, 128, 64, 6, 3, 3, 0);
  for (int i = 0; i < 3; ++i) {
    Wdr1[i] = pack(20 + 2 * i, 64, 64, 128, 7, 3, 3, 0);
    Wdr2[i] = pack(21 + 2 * i, 128, 128, 64, 6, 1, 1, 0);
  }
  __bf16* Wdt1 = pack(26, 128, 128, 128, 7, 4, 4, 1);
  __bf16* Wdt2 = pack(28, 64, 64, 128, 7, 4, 4, 1);
  __bf16* Wdt3 = pack(30, 3, 16, 64, 6, 4, 4, 1);

  const float* B = nullptr;
  auto bias = [&](int pi) { return (const float*)d_in[pi]; };

  float* out0    = (float*)d_out;
  float* rgb_e   = out0 + 1;
  float* ind_out = out0 + 1 + 16 * 3 * 256 * 256;

  hipMemsetAsync(ACC, 0, 2 * sizeof(float), stream);
  pack_input_kernel<<<(16 * 65536 + 255) / 256, 256, 0, stream>>>(rgb, INP);

  // ---- Encoder ----
  conv_launch(stream, P0, INP, Wec1, bias(2), 0, 0, 0, 0, 16, 256, 256, 14, 7,  64,  64, 3, 4, 4, 128,  2, 1, 0, 0, 1, 0);
  conv_launch(stream, P2, P0,  Wec2, bias(4), 0, 0, 0, 0, 16, 128, 128, 12, 6, 128, 128, 6, 4, 4, 1024, 2, 1, 0, 0, 1, 0);
  conv_launch(stream, P3, P2,  Wec3, bias(6), 0, 0, 0, 0, 16,  64,  64, 10, 5, 128, 128, 7, 4, 4, 2048, 2, 1, 0, 0, 1, 0);
  conv_launch(stream, P4, P3,  Wec4, bias(8), 0, 0, 0, 0, 16,  32,  32, 10, 5, 128, 128, 7, 3, 3, 1152, 1, 1, 0, 0, 0, 0);
  for (int i = 0; i < 3; ++i) {
    conv_launch(stream, P5, P4, Wer1[i], B, 0,  0, 0, 0, 16, 32, 32, 10, 5,  64,  64, 7, 3, 3, 1152, 1, 1, 0, 1, 0, 0);
    conv_launch(stream, P4, P5, Wer2[i], B, P4, 0, 0, 0, 16, 32, 32, 10, 5, 128, 128, 6, 1, 1,   64, 1, 0, 0, 1, 0, 0);
  }
  conv_launch(stream, P6, P4, Wpvq, bias(16), 0, 0, 0, 0, 16, 32, 32, 10, 5, 64, 64, 7, 1, 1, 128, 1, 0, 0, 1, 0, 0);

  // ---- Vector quantization ----
  code_norms_kernel<<<2, 256, 0, stream>>>(cbk, NRM);
  conv_launch(stream, SCR, P6, Wcbk, B, 0, NRM, 0, 0, 16, 32, 32, 10, 5, 512, 512, 6, 1, 1, 64, 1, 0, 0, 0, 0, 1);
  argmin_kernel<<<16384, 32, 0, stream>>>(SCR, IND, ind_out);
  gather_latent_kernel<<<16384, 64, 0, stream>>>(P6, cbk, IND, P7, ACC + 1);

  // ---- Decoder ----
  conv_launch(stream, P3, P7, Wdc1, bias(19), 0, 0, 0, 0, 16, 32, 32, 10, 5, 128, 128, 6, 3, 3, 576, 1, 1, 0, 0, 0, 0);
  for (int i = 0; i < 3; ++i) {
    conv_launch(stream, P5, P3, Wdr1[i], B, 0,  0, 0, 0, 16, 32, 32, 10, 5,  64,  64, 7, 3, 3, 1152, 1, 1, 0, 1, 0, 0);
    conv_launch(stream, P3, P5, Wdr2[i], B, P3, 0, 0, 0, 16, 32, 32, 10, 5, 128, 128, 6, 1, 1,   64, 1, 0, 0, 1, 0, 0);
  }
  // Transposed convs: dilated-conv equivalent, pad = k-1-p = 2.
  conv_launch(stream, P2, P3, Wdt1, bias(27), 0, 0, 0, 0, 16, 32, 32, 12, 6, 128, 128, 7, 4, 4, 2048, 1, 2, 1, 1, 1, 0);
  conv_launch(stream, P0, P2, Wdt2, bias(29), 0, 0, 0, 0, 16, 64, 64, 14, 7,  64,  64, 7, 4, 4, 2048, 1, 2, 1, 0, 1, 0);
  conv_launch(stream, rgb_e, P0, Wdt3, bias(31), 0, 0, rgb, ACC,
              16, 128, 128, 16, 8, 3, 16, 6, 4, 4, 1024, 1, 2, 1, 0, 0, 2);

  finalize_kernel<<<1, 1, 0, stream>>>(ACC, out0);
}
